// MambaLayer_87883620810778
// MI455X (gfx1250) — matrix operand order repaired and checked by reading, not compiled, not run on hardware
//
#include <hip/hip_runtime.h>
#include <hip/hip_bf16.h>

typedef __attribute__((ext_vector_type(16))) _Float16 v16h;
typedef __attribute__((ext_vector_type(8)))  _Float16 v8h;
typedef __attribute__((ext_vector_type(8)))  float    v8f;

#define DIM        768
#define D_STATE    64
#define D_CONV     4
#define D_INNER    1536
#define DT_RANK    48
#define DT_PAD     64
#define BATCH      2
#define SEQ        1024
#define M_TOTAL    (BATCH * SEQ)          // 2048
#define N_XZ       (2 * D_INNER)          // 3072
#define N_XDBL     (DT_RANK + 2*D_STATE)  // 176
#define N_XDBL_PAD 192                    // padded to a multiple of 64

__device__ __forceinline__ float sigmoidf_(float x) { return 1.0f / (1.0f + __expf(-x)); }
__device__ __forceinline__ float siluf_(float x)    { return x * sigmoidf_(x); }

// ---------------------------------------------------------------------------
// LayerNorm -> f16
// ---------------------------------------------------------------------------
__global__ void layernorm_f16_kernel(const float* __restrict__ x,
                                     const float* __restrict__ gamma,
                                     const float* __restrict__ beta,
                                     _Float16* __restrict__ xn)
{
    __shared__ float red[256];
    const int row = blockIdx.x;
    const float* xr = x + (size_t)row * DIM;

    float s = 0.f;
    for (int i = threadIdx.x; i < DIM; i += 256) s += xr[i];
    red[threadIdx.x] = s;
    __syncthreads();
    for (int off = 128; off > 0; off >>= 1) {
        if (threadIdx.x < off) red[threadIdx.x] += red[threadIdx.x + off];
        __syncthreads();
    }
    const float mu = red[0] * (1.0f / DIM);
    __syncthreads();

    float v = 0.f;
    for (int i = threadIdx.x; i < DIM; i += 256) { float d = xr[i] - mu; v += d * d; }
    red[threadIdx.x] = v;
    __syncthreads();
    for (int off = 128; off > 0; off >>= 1) {
        if (threadIdx.x < off) red[threadIdx.x] += red[threadIdx.x + off];
        __syncthreads();
    }
    const float rstd = rsqrtf(red[0] * (1.0f / DIM) + 1e-5f);

    for (int i = threadIdx.x; i < DIM; i += 256)
        xn[(size_t)row * DIM + i] = (_Float16)((xr[i] - mu) * rstd * gamma[i] + beta[i]);
}

// ---------------------------------------------------------------------------
// weight prep
// ---------------------------------------------------------------------------
__global__ void f32_to_f16_kernel(const float* __restrict__ src, _Float16* __restrict__ dst, int n)
{
    int i = blockIdx.x * blockDim.x + threadIdx.x;
    if (i < n) dst[i] = (_Float16)src[i];
}

// W_dt (1536 x 48) -> f16 padded to (1536 x 64)
__global__ void wdt_pad_kernel(const float* __restrict__ W_dt, _Float16* __restrict__ dst)
{
    int i = blockIdx.x * blockDim.x + threadIdx.x;
    if (i >= D_INNER * DT_PAD) return;
    int r = i % DT_PAD, c = i / DT_PAD;
    dst[i] = (r < DT_RANK) ? (_Float16)W_dt[c * DT_RANK + r] : (_Float16)0.0f;
}

// W_x (176 x 1536) -> f16 padded to (192 x 1536), zero rows 176..191
__global__ void wx_pad_kernel(const float* __restrict__ W_x, _Float16* __restrict__ dst)
{
    int i = blockIdx.x * blockDim.x + threadIdx.x;
    if (i >= N_XDBL_PAD * D_INNER) return;
    int row = i / D_INNER, c = i - row * D_INNER;
    dst[i] = (row < N_XDBL) ? (_Float16)W_x[(size_t)row * D_INNER + c] : (_Float16)0.0f;
}

__global__ void aneg_kernel(const float* __restrict__ A_log, float* __restrict__ Aneg, int n)
{
    int i = blockIdx.x * blockDim.x + threadIdx.x;
    if (i < n) Aneg[i] = -__expf(A_log[i]);
}

// ---------------------------------------------------------------------------
// fragment loaders
// A fragment (16-bit A 16x32): lane(l,hi) holds halves 0..7 = K[k+hi*8 .. +7],
//                              halves 8..15 = K[k+16+hi*8 .. +7]
// B fragment (16-bit B 32x16): same K order as A: halves 0..7 = K[k+hi*8 .. +7],
//                              halves 8..15 = K[k+16+hi*8 .. +7]
// ---------------------------------------------------------------------------
__device__ __forceinline__ v16h load_a_frag(const _Float16* __restrict__ row, int k, int hi)
{
    const v8h lo = *(const v8h*)(row + k + hi * 8);
    const v8h h8 = *(const v8h*)(row + k + 16 + hi * 8);
    v16h f;
#pragma unroll
    for (int t = 0; t < 8; ++t) { f[t] = lo[t]; f[t + 8] = h8[t]; }
    return f;
}

__device__ __forceinline__ v16h load_b_frag(const _Float16* __restrict__ row, int k, int hi)
{
    const _Float16* p = row + k + hi * 8;
    const v8h b0 = *(const v8h*)(p);
    const v8h b1 = *(const v8h*)(p + 16);
    v16h f;
#pragma unroll
    for (int t = 0; t < 8; ++t) { f[t] = b0[t]; f[t + 8] = b1[t]; }
    return f;
}

// ---------------------------------------------------------------------------
// WMMA GEMM: C[M,N] = A[M,K] * B[N,K]^T, fp16 inputs, fp32 accumulate.
// Requirements: K % 64 == 0, M % 32 == 0, N % 64 == 0.
// A and B buffers must have >= 256 bytes of slack after the last row
// (the software-pipelined prefetch overreads up to 64 B on the final row).
// Each wave computes a 32x64 tile: 2 A-frags x 4 B-frags = 8 wmma/half-step,
// double-buffered over K (ping-pong), 16 wmma per 64-wide K step.
// blockDim = 128 (4 waves). grid = (M/32, ceil(N/256)).
// __launch_bounds__(128, 1): allow the full VGPR budget (the ~180-VGPR
// working set spilled fragment pointers to scratch at default occupancy).
// ---------------------------------------------------------------------------
__global__ __launch_bounds__(128, 1)
void wmma_gemm_f16_kernel(const _Float16* __restrict__ A,
                          const _Float16* __restrict__ B,
                          float* __restrict__ C,
                          int M, int N, int K)
{
    const int lane  = threadIdx.x & 31;
    const int wave  = threadIdx.x >> 5;
    const int l     = lane & 15;
    const int hi    = lane >> 4;
    const int m0    = blockIdx.x * 32;
    const int nBase = (blockIdx.y * 4 + wave) * 64;
    // wave-uniform guard (only the tail wave of the N=192 GEMM hits this)
    if (__builtin_amdgcn_readfirstlane(nBase) >= N) return;

    v8f acc[2][4] = {};

    const _Float16* A0  = A + (size_t)(m0 + l) * K;
    const _Float16* A1  = A + (size_t)(m0 + 16 + l) * K;
    const _Float16* Bp0 = B + (size_t)(nBase + 0 * 16 + l) * K;
    const _Float16* Bp1 = B + (size_t)(nBase + 1 * 16 + l) * K;
    const _Float16* Bp2 = B + (size_t)(nBase + 2 * 16 + l) * K;
    const _Float16* Bp3 = B + (size_t)(nBase + 3 * 16 + l) * K;

    // prologue: load "current" set at k = 0
    v16h a0c = load_a_frag(A0, 0, hi);
    v16h a1c = load_a_frag(A1, 0, hi);
    v16h b0c = load_b_frag(Bp0, 0, hi);
    v16h b1c = load_b_frag(Bp1, 0, hi);
    v16h b2c = load_b_frag(Bp2, 0, hi);
    v16h b3c = load_b_frag(Bp3, 0, hi);

    for (int k = 0; k < K; k += 64) {
        // prefetch "next" set at k+32 (overlaps with the 8 wmma below)
        v16h a0n = load_a_frag(A0, k + 32, hi);
        v16h a1n = load_a_frag(A1, k + 32, hi);
        v16h b0n = load_b_frag(Bp0, k + 32, hi);
        v16h b1n = load_b_frag(Bp1, k + 32, hi);
        v16h b2n = load_b_frag(Bp2, k + 32, hi);
        v16h b3n = load_b_frag(Bp3, k + 32, hi);

        acc[0][0] = __builtin_amdgcn_wmma_f32_16x16x32_f16(false, a0c, false, b0c, (short)0, acc[0][0], false, false);
        acc[1][0] = __builtin_amdgcn_wmma_f32_16x16x32_f16(false, a1c, false, b0c, (short)0, acc[1][0], false, false);
        acc[0][1] = __builtin_amdgcn_wmma_f32_16x16x32_f16(false, a0c, false, b1c, (short)0, acc[0][1], false, false);
        acc[1][1] = __builtin_amdgcn_wmma_f32_16x16x32_f16(false, a1c, false, b1c, (short)0, acc[1][1], false, false);
        acc[0][2] = __builtin_amdgcn_wmma_f32_16x16x32_f16(false, a0c, false, b2c, (short)0, acc[0][2], false, false);
        acc[1][2] = __builtin_amdgcn_wmma_f32_16x16x32_f16(false, a1c, false, b2c, (short)0, acc[1][2], false, false);
        acc[0][3] = __builtin_amdgcn_wmma_f32_16x16x32_f16(false, a0c, false, b3c, (short)0, acc[0][3], false, false);
        acc[1][3] = __builtin_amdgcn_wmma_f32_16x16x32_f16(false, a1c, false, b3c, (short)0, acc[1][3], false, false);

        // load "current" set for the next loop iteration at k+64
        // (on the last iteration this overreads into the 256B buffer pad; unused)
        a0c = load_a_frag(A0, k + 64, hi);
        a1c = load_a_frag(A1, k + 64, hi);
        b0c = load_b_frag(Bp0, k + 64, hi);
        b1c = load_b_frag(Bp1, k + 64, hi);
        b2c = load_b_frag(Bp2, k + 64, hi);
        b3c = load_b_frag(Bp3, k + 64, hi);

        acc[0][0] = __builtin_amdgcn_wmma_f32_16x16x32_f16(false, a0n, false, b0n, (short)0, acc[0][0], false, false);
        acc[1][0] = __builtin_amdgcn_wmma_f32_16x16x32_f16(false, a1n, false, b0n, (short)0, acc[1][0], false, false);
        acc[0][1] = __builtin_amdgcn_wmma_f32_16x16x32_f16(false, a0n, false, b1n, (short)0, acc[0][1], false, false);
        acc[1][1] = __builtin_amdgcn_wmma_f32_16x16x32_f16(false, a1n, false, b1n, (short)0, acc[1][1], false, false);
        acc[0][2] = __builtin_amdgcn_wmma_f32_16x16x32_f16(false, a0n, false, b2n, (short)0, acc[0][2], false, false);
        acc[1][2] = __builtin_amdgcn_wmma_f32_16x16x32_f16(false, a1n, false, b2n, (short)0, acc[1][2], false, false);
        acc[0][3] = __builtin_amdgcn_wmma_f32_16x16x32_f16(false, a0n, false, b3n, (short)0, acc[0][3], false, false);
        acc[1][3] = __builtin_amdgcn_wmma_f32_16x16x32_f16(false, a1n, false, b3n, (short)0, acc[1][3], false, false);
    }

#pragma unroll
    for (int j = 0; j < 2; ++j) {
        const int rowBase = m0 + j * 16 + hi * 8;
#pragma unroll
        for (int i = 0; i < 4; ++i) {
            const int col = nBase + i * 16 + l;
#pragma unroll
            for (int r = 0; r < 8; ++r)
                C[(size_t)(rowBase + r) * N + col] = acc[j][i][r];
        }
    }
}

// ---------------------------------------------------------------------------
// causal depthwise conv (k=4) + SiLU; reads u half of xz in-place (ld = N_XZ)
// ---------------------------------------------------------------------------
__global__ void conv_silu_kernel(const float* __restrict__ xz,
                                 const float* __restrict__ conv_w,
                                 const float* __restrict__ conv_b,
                                 float* __restrict__ u_f32,
                                 _Float16* __restrict__ u_h)
{
    int idx = blockIdx.x * blockDim.x + threadIdx.x;
    if (idx >= M_TOTAL * D_INNER) return;
    const int c = idx % D_INNER;
    const int m = idx / D_INNER;
    const int b = m / SEQ;
    const int l = m % SEQ;

    float acc = conv_b[c];
#pragma unroll
    for (int j = 0; j < D_CONV; ++j) {
        const int ll = l - (D_CONV - 1) + j;
        if (ll >= 0)
            acc += conv_w[c * D_CONV + j] * xz[(size_t)(b * SEQ + ll) * N_XZ + c];
    }
    const float s = siluf_(acc);
    u_f32[idx] = s;
    u_h[idx]   = (_Float16)s;
}

// ---------------------------------------------------------------------------
// dt slice of x_dbl (ld = N_XDBL_PAD) -> f16, K padded 48 -> 64
// ---------------------------------------------------------------------------
__global__ void dt_pad_kernel(const float* __restrict__ x_dbl, _Float16* __restrict__ dt_h)
{
    int i = blockIdx.x * blockDim.x + threadIdx.x;
    if (i >= M_TOTAL * DT_PAD) return;
    int r = i % DT_PAD, m = i / DT_PAD;
    dt_h[i] = (r < DT_RANK) ? (_Float16)x_dbl[(size_t)m * N_XDBL_PAD + r] : (_Float16)0.0f;
}

// ---------------------------------------------------------------------------
// selective scan: one wave per (batch, channel), 2 states per lane (wave32)
// ---------------------------------------------------------------------------
__global__ void scan_kernel(const float* __restrict__ delta_pre,
                            const float* __restrict__ b_dt,
                            const float* __restrict__ u,
                            const float* __restrict__ x_dbl,
                            const float* __restrict__ Aneg,
                            const float* __restrict__ Dv,
                            float* __restrict__ y)
{
    const int gid  = blockIdx.x * blockDim.x + threadIdx.x;
    const int wid  = gid >> 5;
    const int lane = threadIdx.x & 31;
    if (wid >= BATCH * D_INNER) return;
    const int b = wid / D_INNER;
    const int c = wid - b * D_INNER;

    const float a0  = Aneg[c * D_STATE + 2 * lane];
    const float a1  = Aneg[c * D_STATE + 2 * lane + 1];
    const float Dc  = Dv[c];
    const float bdt = b_dt[c];

    float h0 = 0.f, h1 = 0.f;
    for (int l = 0; l < SEQ; ++l) {
        const int m = b * SEQ + l;
        float dp = delta_pre[(size_t)m * D_INNER + c] + bdt;
        const float d  = (dp > 20.f) ? dp : log1pf(__expf(dp));   // softplus
        const float uu = u[(size_t)m * D_INNER + c];
        const float2 Bv = *(const float2*)(x_dbl + (size_t)m * N_XDBL_PAD + DT_RANK + 2 * lane);
        const float2 Cv = *(const float2*)(x_dbl + (size_t)m * N_XDBL_PAD + DT_RANK + D_STATE + 2 * lane);

        const float du = d * uu;
        h0 = __expf(d * a0) * h0 + du * Bv.x;
        h1 = __expf(d * a1) * h1 + du * Bv.y;

        float p = h0 * Cv.x + h1 * Cv.y;
#pragma unroll
        for (int off = 16; off > 0; off >>= 1) p += __shfl_xor(p, off, 32);
        if (lane == 0) y[(size_t)m * D_INNER + c] = p + Dc * uu;
    }
}

// ---------------------------------------------------------------------------
// y * silu(z) -> f16
// ---------------------------------------------------------------------------
__global__ void ygate_kernel(const float* __restrict__ y,
                             const float* __restrict__ xz,
                             _Float16* __restrict__ ymul)
{
    int idx = blockIdx.x * blockDim.x + threadIdx.x;
    if (idx >= M_TOTAL * D_INNER) return;
    const int c = idx % D_INNER;
    const int m = idx / D_INNER;
    const float z = xz[(size_t)m * N_XZ + D_INNER + c];
    ymul[idx] = (_Float16)(y[idx] * siluf_(z));
}

// ---------------------------------------------------------------------------
// launch
// ---------------------------------------------------------------------------
extern "C" void kernel_launch(void* const* d_in, const int* in_sizes, int n_in,
                              void* d_out, int out_size, void* d_ws, size_t ws_size,
                              hipStream_t stream)
{
    const float* x      = (const float*)d_in[0];
    const float* gamma  = (const float*)d_in[1];
    const float* beta   = (const float*)d_in[2];
    const float* W_in   = (const float*)d_in[3];
    const float* conv_w = (const float*)d_in[4];
    const float* conv_b = (const float*)d_in[5];
    const float* W_x    = (const float*)d_in[6];
    const float* W_dt   = (const float*)d_in[7];
    const float* b_dt   = (const float*)d_in[8];
    const float* A_log  = (const float*)d_in[9];
    const float* Dvec   = (const float*)d_in[10];
    const float* W_out  = (const float*)d_in[11];
    float* out = (float*)d_out;

    // workspace carve-up (256B aligned; f16 GEMM operands get 256B slack for
    // the pipelined prefetch overread)
    char* ws = (char*)d_ws;
    size_t off = 0;
    auto carve = [&](size_t bytes) -> char* {
        char* p = ws + off;
        off = (off + bytes + 255) & ~(size_t)255;
        return p;
    };
    _Float16* xn_h      = (_Float16*)carve((size_t)M_TOTAL * DIM * 2 + 256);
    _Float16* w_in_h    = (_Float16*)carve((size_t)N_XZ * DIM * 2 + 256);
    float*    xz        = (float*)   carve((size_t)M_TOTAL * N_XZ * 4);
    float*    u_f32     = (float*)   carve((size_t)M_TOTAL * D_INNER * 4);
    _Float16* u_h       = (_Float16*)carve((size_t)M_TOTAL * D_INNER * 2 + 256);
    _Float16* w_x_h     = (_Float16*)carve((size_t)N_XDBL_PAD * D_INNER * 2 + 256);
    float*    x_dbl     = (float*)   carve((size_t)M_TOTAL * N_XDBL_PAD * 4);
    _Float16* dt_h      = (_Float16*)carve((size_t)M_TOTAL * DT_PAD * 2 + 256);
    _Float16* w_dt_h    = (_Float16*)carve((size_t)D_INNER * DT_PAD * 2 + 256);
    float*    delta_pre = (float*)   carve((size_t)M_TOTAL * D_INNER * 4);
    float*    Aneg      = (float*)   carve((size_t)D_INNER * D_STATE * 4);
    float*    y_f32     = (float*)   carve((size_t)M_TOTAL * D_INNER * 4);
    _Float16* ymul_h    = (_Float16*)carve((size_t)M_TOTAL * D_INNER * 2 + 256);
    _Float16* w_out_h   = (_Float16*)carve((size_t)DIM * D_INNER * 2 + 256);

    const int TB = 256;
    auto blocks = [](int n, int tb) { return (n + tb - 1) / tb; };

    // 1. layernorm -> f16
    layernorm_f16_kernel<<<M_TOTAL, TB, 0, stream>>>(x, gamma, beta, xn_h);

    // 2. weight prep
    f32_to_f16_kernel<<<blocks(N_XZ * DIM, TB), TB, 0, stream>>>(W_in, w_in_h, N_XZ * DIM);
    f32_to_f16_kernel<<<blocks(DIM * D_INNER, TB), TB, 0, stream>>>(W_out, w_out_h, DIM * D_INNER);
    wx_pad_kernel<<<blocks(N_XDBL_PAD * D_INNER, TB), TB, 0, stream>>>(W_x, w_x_h);
    wdt_pad_kernel<<<blocks(D_INNER * DT_PAD, TB), TB, 0, stream>>>(W_dt, w_dt_h);
    aneg_kernel<<<blocks(D_INNER * D_STATE, TB), TB, 0, stream>>>(A_log, Aneg, D_INNER * D_STATE);

    // 3. xz = xn @ W_in^T      (2048 x 768 x 3072)
    wmma_gemm_f16_kernel<<<dim3(M_TOTAL / 32, (N_XZ + 255) / 256), 128, 0, stream>>>(
        xn_h, w_in_h, xz, M_TOTAL, N_XZ, DIM);

    // 4. conv4 + silu -> u
    conv_silu_kernel<<<blocks(M_TOTAL * D_INNER, TB), TB, 0, stream>>>(xz, conv_w, conv_b, u_f32, u_h);

    // 5. x_dbl = u @ W_x_pad^T (2048 x 1536 x 192)
    wmma_gemm_f16_kernel<<<dim3(M_TOTAL / 32, (N_XDBL_PAD + 255) / 256), 128, 0, stream>>>(
        u_h, w_x_h, x_dbl, M_TOTAL, N_XDBL_PAD, D_INNER);

    // 6. dt slice, padded to K=64
    dt_pad_kernel<<<blocks(M_TOTAL * DT_PAD, TB), TB, 0, stream>>>(x_dbl, dt_h);

    // 7. delta_pre = dt @ W_dt^T  (2048 x 64 x 1536); softplus folded into scan
    wmma_gemm_f16_kernel<<<dim3(M_TOTAL / 32, (D_INNER + 255) / 256), 128, 0, stream>>>(
        dt_h, w_dt_h, delta_pre, M_TOTAL, D_INNER, DT_PAD);

    // 8. selective scan: 3072 waves
    {
        const int waves = BATCH * D_INNER;
        scan_kernel<<<blocks(waves * 32, 128), 128, 0, stream>>>(
            delta_pre, b_dt, u_f32, x_dbl, Aneg, Dvec, y_f32);
    }

    // 9. gate
    ygate_kernel<<<blocks(M_TOTAL * D_INNER, TB), TB, 0, stream>>>(y_f32, xz, ymul_h);

    // 10. out = (y*silu(z)) @ W_out^T  (2048 x 1536 x 768) -> d_out
    wmma_gemm_f16_kernel<<<dim3(M_TOTAL / 32, (DIM + 255) / 256), 128, 0, stream>>>(
        ymul_h, w_out_h, out, M_TOTAL, DIM, D_INNER);

    (void)in_sizes; (void)n_in; (void)out_size; (void)ws_size;
}
